// Scale_dot_Attention_40183714021645
// MI455X (gfx1250) — matrix operand order repaired and checked by reading, not compiled, not run on hardware
//
#include <hip/hip_runtime.h>

typedef __attribute__((ext_vector_type(16))) __bf16 v16bf;
typedef __attribute__((ext_vector_type(8)))  __bf16 v8bf;
typedef __attribute__((ext_vector_type(4)))  __bf16 v4bf;
typedef __attribute__((ext_vector_type(8)))  float  v8f;

#define WMMA_BF16(A,B,C) __builtin_amdgcn_wmma_f32_16x16x32_bf16( \
    false, (A), false, (B), (short)0, (C), false, false)

#define CAT16(LO,HI) __builtin_shufflevector((LO), (HI), 0,1,2,3,4,5,6,7,8,9,10,11,12,13,14,15)

constexpr int BB = 8, HH = 8, SS = 1024, DD = 64;
constexpr float SCALE = 0.125f;            // 1/sqrt(64), folded into Q fragment
constexpr float NEGV  = -1000000000.0f;

__global__ __launch_bounds__(256)
void attn_fused_kernel(const float* __restrict__ Q, const float* __restrict__ K,
                       const float* __restrict__ V, const int* __restrict__ mask,
                       const float* __restrict__ adj,
                       float* __restrict__ out_res, float* __restrict__ out_p)
{
    __shared__ __bf16 Klds[128][72];    // 128 keys x 64 d, stride 72 (conflict-free)
    __shared__ __bf16 Vt[64][136];      // 64 d x 128 keys (transposed), stride 136
    __shared__ __bf16 Pst[8][16][40];   // per-wave P staging, stride 40

    const int tid  = threadIdx.x;
    const int lane = tid & 31;
    const int wave = tid >> 5;
    const int hi   = lane >> 4;
    const int ln   = lane & 15;

    const int qblk = blockIdx.x & 7;
    const int bh   = blockIdx.x >> 3;
    const int b    = bh >> 3;
    const int q0   = qblk * 128 + wave * 16;

    // uniform bases; hot accesses use base + 32-bit offset (saddr+voffset form)
    const float* Qp = Q + (size_t)bh * SS * DD;
    const float* Kp = K + (size_t)bh * SS * DD;
    const float* Vp = V + (size_t)bh * SS * DD;
    const int*   Mb = mask + (size_t)b * SS * SS;
    const float* Ab = adj  + (size_t)b * (SS-1) * (SS-1);
    float*       Pb = out_p   + (size_t)bh * SS * SS;
    float*       Ob = out_res + (size_t)bh * SS * DD;

    // cooperative chunk loaders (registers only; stores happen separately)
    auto loadK = [&](int ch, float4* fk) {
        #pragma unroll
        for (int it = 0; it < 8; ++it) {
            const int idx = (it*256 + tid) * 4;
            fk[it] = *(const float4*)(Kp + (unsigned)((ch*128 + (idx >> 6)) * DD + (idx & 63)));
        }
    };
    auto loadV = [&](int ch, float4* fv) {
        #pragma unroll
        for (int it = 0; it < 8; ++it) {
            const int idx = (it*256 + tid) * 4;
            fv[it] = *(const float4*)(Vp + (unsigned)((ch*128 + (idx >> 6)) * DD + (idx & 63)));
        }
    };
    auto storeK = [&](const float4* fk) {
        #pragma unroll
        for (int it = 0; it < 8; ++it) {
            const int idx = (it*256 + tid) * 4;
            v4bf kb; kb[0]=(__bf16)fk[it].x; kb[1]=(__bf16)fk[it].y;
                     kb[2]=(__bf16)fk[it].z; kb[3]=(__bf16)fk[it].w;
            *(v4bf*)&Klds[idx >> 6][idx & 63] = kb;
        }
    };
    auto storeV = [&](const float4* fv) {
        #pragma unroll
        for (int it = 0; it < 8; ++it) {
            const int idx = (it*256 + tid) * 4;
            const int row = idx >> 6, col = idx & 63;
            Vt[col+0][row] = (__bf16)fv[it].x;
            Vt[col+1][row] = (__bf16)fv[it].y;
            Vt[col+2][row] = (__bf16)fv[it].z;
            Vt[col+3][row] = (__bf16)fv[it].w;
        }
    };

    // ---- Q A-fragments (two K=32 chunks over D=64), scale folded in ----
    const unsigned qoff = (unsigned)(q0 + ln) * DD;
    v16bf a0, a1;
    #pragma unroll
    for (int i = 0; i < 8; ++i) {
        a0[i]   = (__bf16)(Qp[qoff +      hi*16 + i] * SCALE);
        a0[i+8] = (__bf16)(Qp[qoff +  8 + hi*16 + i] * SCALE);
        a1[i]   = (__bf16)(Qp[qoff + 32 + hi*16 + i] * SCALE);
        a1[i+8] = (__bf16)(Qp[qoff + 40 + hi*16 + i] * SCALE);
    }

    // per-row state (constant-indexed)
    unsigned moff[8], aoff[8], poff[8];
    bool     qedge[8];
    float    mrow[8], lrow[8];
    unsigned long long okb[8];
    #pragma unroll
    for (int r = 0; r < 8; ++r) {
        const int qr = q0 + r + hi*8;
        moff[r]  = (unsigned)qr * SS;
        aoff[r]  = (unsigned)((qr < SS-2) ? qr : (SS-2)) * (SS-1);
        poff[r]  = (unsigned)qr * SS;
        qedge[r] = (qr == SS-1);
        mrow[r]  = -__builtin_inff();
        lrow[r]  = 0.f;
        okb[r]   = 0ull;
    }

    // ================= PASS 1: masked scores -> online (max, sumexp) =================
    float4 fk[8];
    loadK(0, fk);
    #pragma unroll 1
    for (int ch = 0; ch < 8; ++ch) {
        storeK(fk);
        __syncthreads();

        float4 fkn[8];
        if (ch < 7) loadK(ch + 1, fkn);               // prefetch overlaps compute below

        const int kbase = ch * 128;
        for (int tt = 0; tt < 8; ++tt) {
            const int kt = ch*8 + tt;
            const int kl = tt*16 + ln;
            v16bf b0 = CAT16(*(const v8bf*)&Klds[kl][     hi*16],
                             *(const v8bf*)&Klds[kl][     hi*16 + 8]);
            v16bf b1 = CAT16(*(const v8bf*)&Klds[kl][32 + hi*16],
                             *(const v8bf*)&Klds[kl][32 + hi*16 + 8]);
            v8f c = {};
            c = WMMA_BF16(a0, b0, c);
            c = WMMA_BF16(a1, b1, c);

            const int  kc    = kbase + tt*16 + ln;
            const int  kcc   = (kc < SS-2) ? kc : (SS-2);
            const bool kedge = (kc == SS-1);
            int mv[8]; float av[8];
            #pragma unroll
            for (int r = 0; r < 8; ++r) { mv[r] = Mb[moff[r] + kc]; av[r] = Ab[aoff[r] + kcc]; }
            #pragma unroll
            for (int r = 0; r < 8; ++r) {
                const bool ok = (mv[r] != 0) & (qedge[r] | kedge | (av[r] > 0.f));
                okb[r] |= (unsigned long long)ok << kt;
                const float s  = ok ? c[r] : NEGV;
                const float d  = s - mrow[r];
                const bool  gt = d > 0.f;
                const float e  = __expf(-__builtin_fabsf(d));
                mrow[r] = gt ? s : mrow[r];
                lrow[r] = gt ? __builtin_fmaf(lrow[r], e, 1.f) : (lrow[r] + e);
            }
        }
        __syncthreads();
        if (ch < 7) {
            #pragma unroll
            for (int it = 0; it < 8; ++it) fk[it] = fkn[it];
        }
    }
    // combine across the 16 lanes holding each row
    #pragma unroll
    for (int r = 0; r < 8; ++r) {
        #pragma unroll
        for (int off = 1; off < 16; off <<= 1) {
            const float mo2 = __shfl_xor(mrow[r], off, 32);
            const float lo2 = __shfl_xor(lrow[r], off, 32);
            const float d   = mo2 - mrow[r];
            const bool  gt  = d > 0.f;
            const float e   = __expf(-__builtin_fabsf(d));
            lrow[r] = gt ? __builtin_fmaf(lrow[r], e, lo2) : __builtin_fmaf(lo2, e, lrow[r]);
            mrow[r] = gt ? mo2 : mrow[r];
        }
        lrow[r] = 1.0f / lrow[r];        // inverse row sum
    }

    // ================= PASS 2: write p_attn, accumulate O = P·V =================
    v8f o0 = {}, o1 = {}, o2 = {}, o3 = {};
    float4 fv[8];
    loadK(0, fk);
    loadV(0, fv);
    #pragma unroll 1
    for (int ch = 0; ch < 8; ++ch) {
        storeK(fk);
        storeV(fv);
        __syncthreads();

        float4 fkn[8], fvn[8];
        if (ch < 7) { loadK(ch + 1, fkn); loadV(ch + 1, fvn); }   // prefetch

        const int kbase = ch * 128;
        for (int s3 = 0; s3 < 4; ++s3) {              // 32 keys per step
            #pragma unroll
            for (int j = 0; j < 2; ++j) {
                const int kt = ch*8 + s3*2 + j;
                const int kl = s3*32 + j*16 + ln;
                v16bf b0 = CAT16(*(const v8bf*)&Klds[kl][     hi*16],
                                 *(const v8bf*)&Klds[kl][     hi*16 + 8]);
                v16bf b1 = CAT16(*(const v8bf*)&Klds[kl][32 + hi*16],
                                 *(const v8bf*)&Klds[kl][32 + hi*16 + 8]);
                v8f c = {};
                c = WMMA_BF16(a0, b0, c);
                c = WMMA_BF16(a1, b1, c);
                const int kc = kbase + s3*32 + j*16 + ln;
                #pragma unroll
                for (int r = 0; r < 8; ++r) {
                    const bool ok = (unsigned)(okb[r] >> kt) & 1u;
                    const float s = ok ? c[r] : NEGV;
                    const float p = __expf(s - mrow[r]) * lrow[r];
                    Pb[poff[r] + kc] = p;             // coalesced 64B runs
                    Pst[wave][r + hi*8][j*16 + ln] = (__bf16)p;
                }
            }
            __builtin_amdgcn_wave_barrier();
            asm volatile("s_wait_dscnt 0" ::: "memory");   // same-wave LDS RAW fence

            const v8bf plo = *(const v8bf*)&Pst[wave][ln][hi*16];
            const v8bf phi = *(const v8bf*)&Pst[wave][ln][hi*16 + 8];
            const v16bf ap = CAT16(plo, phi);
            const int kvloc = s3*32 + hi*16;
            v16bf bv0 = CAT16(*(const v8bf*)&Vt[ 0 + ln][kvloc], *(const v8bf*)&Vt[ 0 + ln][kvloc + 8]);
            v16bf bv1 = CAT16(*(const v8bf*)&Vt[16 + ln][kvloc], *(const v8bf*)&Vt[16 + ln][kvloc + 8]);
            v16bf bv2 = CAT16(*(const v8bf*)&Vt[32 + ln][kvloc], *(const v8bf*)&Vt[32 + ln][kvloc + 8]);
            v16bf bv3 = CAT16(*(const v8bf*)&Vt[48 + ln][kvloc], *(const v8bf*)&Vt[48 + ln][kvloc + 8]);
            o0 = WMMA_BF16(ap, bv0, o0);
            o1 = WMMA_BF16(ap, bv1, o1);
            o2 = WMMA_BF16(ap, bv2, o2);
            o3 = WMMA_BF16(ap, bv3, o3);
            __builtin_amdgcn_wave_barrier();
        }
        __syncthreads();
        if (ch < 7) {
            #pragma unroll
            for (int it = 0; it < 8; ++it) { fk[it] = fkn[it]; fv[it] = fvn[it]; }
        }
    }

    // ---- write O (B,H,S,D) ----
    #pragma unroll
    for (int r = 0; r < 8; ++r) {
        const unsigned oo = (unsigned)(q0 + r + hi*8) * DD + ln;
        Ob[oo]      = o0[r];
        Ob[oo + 16] = o1[r];
        Ob[oo + 32] = o2[r];
        Ob[oo + 48] = o3[r];
    }
}

extern "C" void kernel_launch(void* const* d_in, const int* in_sizes, int n_in,
                              void* d_out, int out_size, void* d_ws, size_t ws_size,
                              hipStream_t stream)
{
    const float* Q  = (const float*)d_in[0];
    const float* K  = (const float*)d_in[1];
    const float* V  = (const float*)d_in[2];
    const int*   M  = (const int*)d_in[3];
    const float* A  = (const float*)d_in[4];

    float* out_res = (float*)d_out;                                   // (B,H,S,D)
    float* out_p   = out_res + (size_t)BB * HH * SS * DD;             // (B,H,S,S)

    attn_fused_kernel<<<dim3(BB * HH * (SS / 128)), 256, 0, stream>>>(
        Q, K, V, M, A, out_res, out_p);
}
